// TSTransformer_14766097563934
// MI455X (gfx1250) — matrix operand order repaired and checked by reading, not compiled, not run on hardware
//
#include <hip/hip_runtime.h>
#include <hip/hip_bf16.h>
#include <stdint.h>

// ---------------- Types for CDNA5 WMMA / TDM ----------------
typedef __bf16 bf16;
typedef __attribute__((ext_vector_type(16))) __bf16        v16bf;
typedef __attribute__((ext_vector_type(8)))  float         v8f;
typedef __attribute__((ext_vector_type(8)))  unsigned int  v8u;
typedef __attribute__((ext_vector_type(4)))  unsigned int  v4u;
typedef __attribute__((ext_vector_type(8)))  int           v8i;
typedef __attribute__((ext_vector_type(4)))  int           v4i;

#define L_   197
#define BT_  128
#define D_   768
#define N_   32
#define NH_  12
#define HD_  64
#define TOK  (L_ * BT_)   // 25216

#if defined(__AMDGCN__) && __has_builtin(__builtin_amdgcn_tensor_load_to_lds)
#define USE_TDM 1
#else
#define USE_TDM 0
#endif

static __device__ __forceinline__ v8f wmma_bf16(v16bf a, v16bf b, v8f c) {
  return __builtin_amdgcn_wmma_f32_16x16x32_bf16(false, a, false, b, (short)0, c, false, false);
}
static __device__ __forceinline__ v8f v8f_zero() {
  v8f z;
#pragma unroll
  for (int i = 0; i < 8; ++i) z[i] = 0.f;
  return z;
}
static __device__ __forceinline__ float wave16_max(float v) {
#pragma unroll
  for (int o = 8; o; o >>= 1) v = fmaxf(v, __shfl_xor(v, o, 32));
  return v;
}
static __device__ __forceinline__ float wave16_sum(float v) {
#pragma unroll
  for (int o = 8; o; o >>= 1) v += __shfl_xor(v, o, 32);
  return v;
}

#if USE_TDM
typedef __attribute__((address_space(3))) void* lds_vptr_t;
static __device__ __forceinline__ uint32_t lds_offset(const void* p) {
  return (uint32_t)(uintptr_t)(lds_vptr_t)(void*)p;
}
// Issue a TDM 2D tile load: tile is tile_rows x tile_cols bf16 elements, packed
// densely into LDS at lds_off; global rows have stride_elems bf16 elements.
// Rows >= tensor_rows are zero-filled by the TDM OOB rule.
static __device__ __forceinline__ void tdm_load_2d(const bf16* gptr, uint32_t lds_off,
                                                   uint32_t tensor_rows, uint32_t tile_rows,
                                                   uint32_t tile_cols, uint32_t stride_elems) {
  uint64_t ga = (uint64_t)(uintptr_t)gptr;
  v4u g0;
  g0[0] = 1u;                                          // count=1 (valid user D#)
  g0[1] = lds_off;                                     // lds_addr
  g0[2] = (uint32_t)(ga & 0xffffffffu);                // global_addr[31:0]
  g0[3] = (uint32_t)((ga >> 32) & 0x01ffffffu) | (2u << 30);  // addr[56:32] | type=2
  v8i g1;
  g1[0] = (int)(1u << 16);                             // workgroup_mask=0, data_size=1 (2B)
  g1[1] = (int)((stride_elems & 0xffffu) << 16);       // tensor_dim0[15:0]
  g1[2] = (int)(((stride_elems >> 16) & 0xffffu) | ((tensor_rows & 0xffffu) << 16));
  g1[3] = (int)(((tensor_rows >> 16) & 0xffffu) | ((tile_cols & 0xffffu) << 16));  // tile_dim0
  g1[4] = (int)(tile_rows & 0xffffu);                  // tile_dim1 (tile_dim2=0)
  g1[5] = (int)stride_elems;                           // tensor_dim0_stride[31:0]
  g1[6] = 0;                                           // stride[47:32] | dim1_stride lo
  g1[7] = 0;
  v4i z4a; v4i z4b;
#pragma unroll
  for (int i = 0; i < 4; ++i) { z4a[i] = 0; z4b[i] = 0; }
#if defined(__clang_major__) && (__clang_major__ >= 23)
  v8i z8;
#pragma unroll
  for (int i = 0; i < 8; ++i) z8[i] = 0;
  __builtin_amdgcn_tensor_load_to_lds(g0, g1, z4a, z4b, z8, 0);
#else
  __builtin_amdgcn_tensor_load_to_lds(g0, g1, z4a, z4b, 0);
#endif
}
#endif  // USE_TDM

// ---------------- fp32 -> bf16 convert ----------------
__global__ void k_cvt_bf16(const float* __restrict__ in, bf16* __restrict__ out, int n) {
  int i = blockIdx.x * 256 + threadIdx.x;
  if (i < n) out[i] = (bf16)in[i];
}

// ---------------- LayerNorm (D=768), wave per row, bf16 out ----------------
__global__ __launch_bounds__(256)
void k_layernorm(const float* __restrict__ x, const float* __restrict__ g,
                 const float* __restrict__ b, bf16* __restrict__ out, int rows) {
  int w = threadIdx.x >> 5, lane = threadIdx.x & 31;
  int row = blockIdx.x * 8 + w;
  if (row >= rows) return;
  const float* xr = x + (size_t)row * D_;
  float vals[24];
  float s = 0.f;
#pragma unroll
  for (int i = 0; i < 24; ++i) { vals[i] = xr[lane + i * 32]; s += vals[i]; }
#pragma unroll
  for (int o = 16; o; o >>= 1) s += __shfl_xor(s, o, 32);
  float mean = s * (1.f / D_);
  float vs = 0.f;
#pragma unroll
  for (int i = 0; i < 24; ++i) { float d = vals[i] - mean; vs += d * d; }
#pragma unroll
  for (int o = 16; o; o >>= 1) vs += __shfl_xor(vs, o, 32);
  float inv = rsqrtf(vs * (1.f / D_) + 1e-5f);
  bf16* orow = out + (size_t)row * D_;
#pragma unroll
  for (int i = 0; i < 24; ++i) {
    int c = lane + i * 32;
    orow[c] = (bf16)(g[c] * (vals[i] - mean) * inv + b[c]);
  }
}

// ---------------- WMMA GEMM:  C[M,N] = A[M,K] * B[N,K]^T (+bias,+resid,act) ----
// A,B bf16 row-major; bias len N; resid_mode: 0 none, 1 full [M,N], 2 row-bcast
// (resid[(row/128), col], N must be 768); act: 0 none, 1 quick-gelu.
#define TM 128
#define TN 128
#define TK 64

template<int OUT_BF16>
__global__ __launch_bounds__(256)
void k_gemm(const bf16* __restrict__ A, const bf16* __restrict__ B,
            const float* __restrict__ bias, const float* __restrict__ resid,
            void* __restrict__ Cout, int M, int N, int K, int act, int resid_mode) {
  const int m0 = blockIdx.y * TM, n0 = blockIdx.x * TN;
  const int tid = threadIdx.x, lane = tid & 31, w = tid >> 5;
  const int wm = w >> 2, wn = w & 3;          // 2 x 4 wave grid; wave tile 64x32
  const int fr = lane & 15, fc = (lane >> 4) * 16;
  const int KT = K / TK;
  v8f acc[4][2];
#pragma unroll
  for (int i = 0; i < 4; ++i)
#pragma unroll
    for (int j = 0; j < 2; ++j) acc[i][j] = v8f_zero();

#if USE_TDM
  // Double-buffered LDS tiles, filled by the Tensor Data Mover.
  __shared__ alignas(128) bf16 As[2][TM][TK];
  __shared__ alignas(128) bf16 Bs[2][TN][TK];
  const uint32_t ldsA[2] = { lds_offset(&As[0][0][0]), lds_offset(&As[1][0][0]) };
  const uint32_t ldsB[2] = { lds_offset(&Bs[0][0][0]), lds_offset(&Bs[1][0][0]) };
  const bf16* Abase = A + (size_t)m0 * K;
  const bf16* Bbase = B + (size_t)n0 * K;
  const uint32_t rowsA = (uint32_t)((M - m0) < TM ? (M - m0) : TM);
  if (w == 0) {
    tdm_load_2d(Abase, ldsA[0], rowsA, TM, TK, (uint32_t)K);
    tdm_load_2d(Bbase, ldsB[0], TN,    TN, TK, (uint32_t)K);
  }
  int buf = 0;
  for (int kt = 0; kt < KT; ++kt) {
    if (w == 0) __builtin_amdgcn_s_wait_tensorcnt(0);
    __syncthreads();
    if (w == 0 && (kt + 1) < KT) {
      tdm_load_2d(Abase + (size_t)(kt + 1) * TK, ldsA[buf ^ 1], rowsA, TM, TK, (uint32_t)K);
      tdm_load_2d(Bbase + (size_t)(kt + 1) * TK, ldsB[buf ^ 1], TN,    TN, TK, (uint32_t)K);
    }
#pragma unroll
    for (int kk = 0; kk < TK / 32; ++kk) {
      v16bf bfrag[2], afrag[4];
#pragma unroll
      for (int j = 0; j < 2; ++j)
        bfrag[j] = *(const v16bf*)&Bs[buf][wn * 32 + j * 16 + fr][kk * 32 + fc];
#pragma unroll
      for (int i = 0; i < 4; ++i)
        afrag[i] = *(const v16bf*)&As[buf][wm * 64 + i * 16 + fr][kk * 32 + fc];
#pragma unroll
      for (int i = 0; i < 4; ++i)
#pragma unroll
        for (int j = 0; j < 2; ++j)
          acc[i][j] = wmma_bf16(afrag[i], bfrag[j], acc[i][j]);
    }
    __syncthreads();
    buf ^= 1;
  }
#else
  // Fallback: manual VGPR staging (also what the host pass parses).
  __shared__ alignas(32) bf16 As[1][TM][TK];
  __shared__ alignas(32) bf16 Bs[1][TN][TK];
  const int l_row = tid >> 1, l_col = (tid & 1) * 32;   // 128 rows x 64 cols
  int arow = m0 + l_row; if (arow > M - 1) arow = M - 1;
  const bf16* Ap = A + (size_t)arow * K + l_col;
  const bf16* Bp = B + (size_t)(n0 + l_row) * K + l_col;
  const int buf = 0;
  for (int kt = 0; kt < KT; ++kt) {
    v8u a0 = *(const v8u*)(Ap + (size_t)kt * TK);
    v8u a1 = *(const v8u*)(Ap + (size_t)kt * TK + 16);
    v8u b0 = *(const v8u*)(Bp + (size_t)kt * TK);
    v8u b1 = *(const v8u*)(Bp + (size_t)kt * TK + 16);
    *(v8u*)&As[0][l_row][l_col]      = a0;
    *(v8u*)&As[0][l_row][l_col + 16] = a1;
    *(v8u*)&Bs[0][l_row][l_col]      = b0;
    *(v8u*)&Bs[0][l_row][l_col + 16] = b1;
    __syncthreads();
    if (kt + 1 < KT) {
      __builtin_prefetch(Ap + (size_t)(kt + 1) * TK, 0, 1);
      __builtin_prefetch(Bp + (size_t)(kt + 1) * TK, 0, 1);
    }
#pragma unroll
    for (int kk = 0; kk < TK / 32; ++kk) {
      v16bf bfrag[2], afrag[4];
#pragma unroll
      for (int j = 0; j < 2; ++j)
        bfrag[j] = *(const v16bf*)&Bs[buf][wn * 32 + j * 16 + fr][kk * 32 + fc];
#pragma unroll
      for (int i = 0; i < 4; ++i)
        afrag[i] = *(const v16bf*)&As[buf][wm * 64 + i * 16 + fr][kk * 32 + fc];
#pragma unroll
      for (int i = 0; i < 4; ++i)
#pragma unroll
        for (int j = 0; j < 2; ++j)
          acc[i][j] = wmma_bf16(afrag[i], bfrag[j], acc[i][j]);
    }
    __syncthreads();
  }
#endif

  const int fh = lane >> 4;
#pragma unroll
  for (int i = 0; i < 4; ++i)
#pragma unroll
    for (int j = 0; j < 2; ++j)
#pragma unroll
      for (int r = 0; r < 8; ++r) {
        int row = m0 + wm * 64 + i * 16 + r + fh * 8;
        int col = n0 + wn * 32 + j * 16 + fr;
        if (row < M) {
          float v = acc[i][j][r];
          if (bias) v += bias[col];
          if (resid_mode == 1)      v += resid[(size_t)row * N + col];
          else if (resid_mode == 2) v += resid[(size_t)(row >> 7) * N + col];
          if (act == 1) v = v / (1.f + __expf(-1.702f * v));   // x*sigmoid(1.702x)
          if (OUT_BF16) ((bf16*)Cout)[(size_t)row * N + col] = (bf16)v;
          else          ((float*)Cout)[(size_t)row * N + col] = v;
        }
      }
}

// ---------------- Self-attention (Lq=Lk=197), one wave per (b,h,16-q-tile) ----
__global__ __launch_bounds__(32)
void k_attn_self(const bf16* __restrict__ QKV, bf16* __restrict__ O) {
  const int b = blockIdx.x & (BT_ - 1);
  const int h = blockIdx.x >> 7;
  const int q0 = blockIdx.y * 16;
  const int lane = threadIdx.x;
  const int fr = lane & 15, fh = lane >> 4;
  __shared__ alignas(32) bf16 W[16][224];
  for (int i = lane; i < 16 * 16; i += 32) W[i >> 4][208 + (i & 15)] = (bf16)0.f;

  int qrow = q0 + fr; if (qrow > L_ - 1) qrow = L_ - 1;
  const bf16* qp = QKV + ((size_t)qrow * BT_ + b) * 2304 + h * HD_;
  v16bf qa0 = *(const v16bf*)(qp + fh * 16);
  v16bf qa1 = *(const v16bf*)(qp + 32 + fh * 16);

  v8f s[13];
#pragma unroll
  for (int t = 0; t < 13; ++t) {
    int krow = t * 16 + fr; if (krow > L_ - 1) krow = L_ - 1;
    const bf16* kp = QKV + ((size_t)krow * BT_ + b) * 2304 + D_ + h * HD_;
    v16bf kb0 = *(const v16bf*)(kp + fh * 16);
    v16bf kb1 = *(const v16bf*)(kp + 32 + fh * 16);
    v8f z = v8f_zero();
    z = wmma_bf16(qa0, kb0, z);
    z = wmma_bf16(qa1, kb1, z);
    s[t] = z;
  }
#pragma unroll
  for (int t = 0; t < 13; ++t)
#pragma unroll
    for (int r = 0; r < 8; ++r) {
      float v = s[t][r] * 0.125f;
      if (t * 16 + fr >= L_) v = -1e30f;
      s[t][r] = v;
    }
  float mx[8], inv[8];
#pragma unroll
  for (int r = 0; r < 8; ++r) {
    float m = -3.0e38f;
#pragma unroll
    for (int t = 0; t < 13; ++t) m = fmaxf(m, s[t][r]);
    mx[r] = wave16_max(m);
  }
#pragma unroll
  for (int t = 0; t < 13; ++t)
#pragma unroll
    for (int r = 0; r < 8; ++r) s[t][r] = __expf(s[t][r] - mx[r]);
#pragma unroll
  for (int r = 0; r < 8; ++r) {
    float a = 0.f;
#pragma unroll
    for (int t = 0; t < 13; ++t) a += s[t][r];
    inv[r] = 1.f / wave16_sum(a);
  }
  __syncthreads();
#pragma unroll
  for (int t = 0; t < 13; ++t)
#pragma unroll
    for (int r = 0; r < 8; ++r)
      W[r + fh * 8][t * 16 + fr] = (bf16)(s[t][r] * inv[r]);
  __syncthreads();

  v8f o[4];
#pragma unroll
  for (int j = 0; j < 4; ++j) o[j] = v8f_zero();
#pragma unroll
  for (int tt = 0; tt < 7; ++tt) {
    v16bf wa = *(const v16bf*)&W[fr][tt * 32 + fh * 16];
#pragma unroll
    for (int j = 0; j < 4; ++j) {
      v16bf vb;
#pragma unroll
      for (int e = 0; e < 16; ++e) {
        int key = tt * 32 + fh * 16 + e; if (key > L_ - 1) key = L_ - 1;
        vb[e] = QKV[((size_t)key * BT_ + b) * 2304 + 2 * D_ + h * HD_ + j * 16 + fr];
      }
      o[j] = wmma_bf16(wa, vb, o[j]);
    }
  }
#pragma unroll
  for (int j = 0; j < 4; ++j)
#pragma unroll
    for (int r = 0; r < 8; ++r) {
      int q = q0 + r + fh * 8;
      if (q < L_)
        O[((size_t)q * BT_ + b) * D_ + h * HD_ + j * 16 + fr] = (bf16)o[j][r];
    }
}

// ---------------- Cross-attention: q = QP (32 rows, shared across b), kv 197 ----
__global__ __launch_bounds__(32)
void k_attn_cross(const bf16* __restrict__ QP, const bf16* __restrict__ KV,
                  const float* __restrict__ mask, bf16* __restrict__ O) {
  const int b = blockIdx.x & (BT_ - 1);
  const int h = blockIdx.x >> 7;
  const int q0 = blockIdx.y * 16;
  const int lane = threadIdx.x;
  const int fr = lane & 15, fh = lane >> 4;
  __shared__ alignas(32) bf16 W[16][224];
  for (int i = lane; i < 16 * 16; i += 32) W[i >> 4][208 + (i & 15)] = (bf16)0.f;

  const bf16* qp = QP + (size_t)(q0 + fr) * D_ + h * HD_;
  v16bf qa0 = *(const v16bf*)(qp + fh * 16);
  v16bf qa1 = *(const v16bf*)(qp + 32 + fh * 16);

  v8f s[13];
#pragma unroll
  for (int t = 0; t < 13; ++t) {
    int krow = t * 16 + fr; if (krow > L_ - 1) krow = L_ - 1;
    const bf16* kp = KV + ((size_t)krow * BT_ + b) * 1536 + h * HD_;
    v16bf kb0 = *(const v16bf*)(kp + fh * 16);
    v16bf kb1 = *(const v16bf*)(kp + 32 + fh * 16);
    v8f z = v8f_zero();
    z = wmma_bf16(qa0, kb0, z);
    z = wmma_bf16(qa1, kb1, z);
    s[t] = z;
  }
#pragma unroll
  for (int t = 0; t < 13; ++t)
#pragma unroll
    for (int r = 0; r < 8; ++r) {
      int col = t * 16 + fr;
      float v = s[t][r] * 0.125f;
      if (col >= L_) v = -1e30f;
      else {
        int qq = q0 + r + fh * 8;   // query index c (< 32)
        float am = (col == 0) ? 1.0f : mask[((size_t)b * N_ + qq) * (L_ - 1) + (col - 1)];
        v += am;
      }
      s[t][r] = v;
    }
  float mx[8], inv[8];
#pragma unroll
  for (int r = 0; r < 8; ++r) {
    float m = -3.0e38f;
#pragma unroll
    for (int t = 0; t < 13; ++t) m = fmaxf(m, s[t][r]);
    mx[r] = wave16_max(m);
  }
#pragma unroll
  for (int t = 0; t < 13; ++t)
#pragma unroll
    for (int r = 0; r < 8; ++r) s[t][r] = __expf(s[t][r] - mx[r]);
#pragma unroll
  for (int r = 0; r < 8; ++r) {
    float a = 0.f;
#pragma unroll
    for (int t = 0; t < 13; ++t) a += s[t][r];
    inv[r] = 1.f / wave16_sum(a);
  }
  __syncthreads();
#pragma unroll
  for (int t = 0; t < 13; ++t)
#pragma unroll
    for (int r = 0; r < 8; ++r)
      W[r + fh * 8][t * 16 + fr] = (bf16)(s[t][r] * inv[r]);
  __syncthreads();

  v8f o[4];
#pragma unroll
  for (int j = 0; j < 4; ++j) o[j] = v8f_zero();
#pragma unroll
  for (int tt = 0; tt < 7; ++tt) {
    v16bf wa = *(const v16bf*)&W[fr][tt * 32 + fh * 16];
#pragma unroll
    for (int j = 0; j < 4; ++j) {
      v16bf vb;
#pragma unroll
      for (int e = 0; e < 16; ++e) {
        int key = tt * 32 + fh * 16 + e; if (key > L_ - 1) key = L_ - 1;
        vb[e] = KV[((size_t)key * BT_ + b) * 1536 + D_ + h * HD_ + j * 16 + fr];
      }
      o[j] = wmma_bf16(wa, vb, o[j]);
    }
  }
#pragma unroll
  for (int j = 0; j < 4; ++j)
#pragma unroll
    for (int r = 0; r < 8; ++r) {
      int q = q0 + r + fh * 8;   // < 32
      O[((size_t)q * BT_ + b) * D_ + h * HD_ + j * 16 + fr] = (bf16)o[j][r];
    }
}

// ---------------- Pooling kernels ----------------
__global__ void k_pool_mask(const float* __restrict__ mask,
                            float* __restrict__ pres, float* __restrict__ rowsum) {
  int b = blockIdx.x, c = threadIdx.x;
  const float* mp = mask + ((size_t)b * N_ + c) * (L_ - 1);
  float s = 0.f;
  for (int t = 0; t < L_ - 1; ++t) s += mp[t];
  rowsum[b * N_ + c] = s;
  pres[b * N_ + c] = (s > 0.f) ? 1.f : 0.f;
}

__global__ void k_counts(const float* __restrict__ pres, const float* __restrict__ rowsum,
                         float* __restrict__ counts) {
  int c = threadIdx.x;
  float s = 0.f;
  for (int b = 0; b < BT_; ++b) s += rowsum[b * N_ + c] + pres[b * N_ + c];
  counts[c] = (s > 1.f) ? s : 1.f;
}

__global__ __launch_bounds__(256)
void k_feat(const float* __restrict__ mask, const float* __restrict__ pres,
            const float* __restrict__ counts, const float* __restrict__ XT,
            float* __restrict__ feat) {
  int c = blockIdx.x;
  int d = blockIdx.y * 256 + threadIdx.x;
  float acc = 0.f;
  for (int b = 0; b < BT_; ++b) {
    acc += pres[b * N_ + c] * XT[(size_t)b * D_ + d];        // CLS token (l=0)
    const float* mp = mask + ((size_t)b * N_ + c) * (L_ - 1);
    for (int t = 0; t < L_ - 1; ++t)
      acc += mp[t] * XT[((size_t)(t + 1) * BT_ + b) * D_ + d];
  }
  feat[(size_t)c * D_ + d] = acc / counts[c];
}

__global__ __launch_bounds__(256)
void k_yout(const float* __restrict__ pres, const float* __restrict__ feat,
            float* __restrict__ yout) {
  int g = blockIdx.x;
  int d = blockIdx.y * 256 + threadIdx.x;
  float num = 0.f, den = 0.f;
  for (int t = 0; t < 8; ++t)
    for (int c = 0; c < N_; ++c) {
      float p = pres[(g * 8 + t) * N_ + c];
      num += p * feat[(size_t)c * D_ + d];
      den += p;
    }
  float v = num / ((den > 1.f) ? den : 1.f);
  for (int t = 0; t < 8; ++t) yout[(size_t)(g * 8 + t) * D_ + d] = v;
}

// ---------------- Host-side orchestration ----------------
struct BlkP {
  const float *wi, *bi, *wo, *bo, *ln1g, *ln1b, *ln2g, *ln2b, *w1, *b1, *w2, *b2;
  bf16 *wi_b, *wo_b, *w1_b, *w2_b;
};

static void cvt(const float* src, bf16* dst, int n, hipStream_t s) {
  k_cvt_bf16<<<(n + 255) / 256, 256, 0, s>>>(src, dst, n);
}
static void ln(const float* x, const float* g, const float* b, bf16* out, int rows, hipStream_t s) {
  k_layernorm<<<(rows + 7) / 8, 256, 0, s>>>(x, g, b, out, rows);
}
static void gemm_f32(const bf16* A, const bf16* B, const float* bias, const float* resid,
                     float* C, int M, int N, int K, int act, int rmode, hipStream_t s) {
  dim3 grid(N / TN, (M + TM - 1) / TM);
  k_gemm<0><<<grid, 256, 0, s>>>(A, B, bias, resid, (void*)C, M, N, K, act, rmode);
}
static void gemm_bf16o(const bf16* A, const bf16* B, const float* bias, const float* resid,
                       bf16* C, int M, int N, int K, int act, int rmode, hipStream_t s) {
  dim3 grid(N / TN, (M + TM - 1) / TM);
  k_gemm<1><<<grid, 256, 0, s>>>(A, B, bias, resid, (void*)C, M, N, K, act, rmode);
}

static void encoder_block(const float* xin, float* xout, const BlkP& P,
                          bf16* H, bf16* SCR, bf16* O, hipStream_t s) {
  ln(xin, P.ln1g, P.ln1b, H, TOK, s);
  gemm_bf16o(H, P.wi_b, P.bi, nullptr, SCR, TOK, 3 * D_, D_, 0, 0, s);       // QKV
  k_attn_self<<<dim3(BT_ * NH_, 13), 32, 0, s>>>(SCR, O);
  gemm_f32(O, P.wo_b, P.bo, xin, xout, TOK, D_, D_, 0, 1, s);                // proj + resid
  ln(xout, P.ln2g, P.ln2b, H, TOK, s);
  gemm_bf16o(H, P.w1_b, P.b1, nullptr, SCR, TOK, 4 * D_, D_, 1, 0, s);       // MLP1 + qgelu
  gemm_f32(SCR, P.w2_b, P.b2, xout, xout, TOK, D_, 4 * D_, 0, 1, s);         // MLP2 + resid
}

extern "C" void kernel_launch(void* const* d_in, const int* in_sizes, int n_in,
                              void* d_out, int out_size, void* d_ws, size_t ws_size,
                              hipStream_t stream) {
  const float* x    = (const float*)d_in[0];
  const float* mask = (const float*)d_in[1];

  BlkP P[3];
  for (int bi = 0; bi < 3; ++bi) {
    auto gp = [&](int j) { return (const float*)d_in[2 + bi * 12 + j]; };
    P[bi].wi = gp(0);  P[bi].bi = gp(1);  P[bi].wo = gp(2);  P[bi].bo = gp(3);
    P[bi].ln1g = gp(4); P[bi].ln1b = gp(5); P[bi].ln2g = gp(6); P[bi].ln2b = gp(7);
    P[bi].w1 = gp(8);  P[bi].b1 = gp(9);  P[bi].w2 = gp(10); P[bi].b2 = gp(11);
  }

  char* wsb = (char*)d_ws;
  size_t off = 0;
  auto carve = [&](size_t bytes) -> void* {
    void* p = wsb + off;
    off += (bytes + 255) & ~(size_t)255;
    return p;
  };
  for (int bi = 0; bi < 3; ++bi) {
    P[bi].wi_b = (bf16*)carve((size_t)3 * D_ * D_ * 2);
    P[bi].wo_b = (bf16*)carve((size_t)D_ * D_ * 2);
    P[bi].w1_b = (bf16*)carve((size_t)4 * D_ * D_ * 2);
    P[bi].w2_b = (bf16*)carve((size_t)4 * D_ * D_ * 2);
  }
  float* X   = (float*)carve((size_t)TOK * D_ * 4);
  float* X2  = (float*)carve((size_t)TOK * D_ * 4);
  float* XT  = (float*)carve((size_t)TOK * D_ * 4);
  bf16*  H   = (bf16*)carve((size_t)TOK * D_ * 2);
  bf16*  SCR = (bf16*)carve((size_t)TOK * 4 * D_ * 2);   // QKV / KV / MLP hidden
  bf16*  O   = (bf16*)carve((size_t)TOK * D_ * 2);
  float* pres   = (float*)carve((size_t)BT_ * N_ * 4);
  float* rowsum = (float*)carve((size_t)BT_ * N_ * 4);
  float* counts = (float*)carve(N_ * 4);
  float* feat   = (float*)carve((size_t)N_ * D_ * 4);
  bf16*  qln    = (bf16*)carve((size_t)N_ * D_ * 2);
  bf16*  qp     = (bf16*)carve((size_t)N_ * D_ * 2);

  // Weight conversion to bf16
  for (int bi = 0; bi < 3; ++bi) {
    cvt(P[bi].wi, P[bi].wi_b, 3 * D_ * D_, stream);
    cvt(P[bi].wo, P[bi].wo_b, D_ * D_, stream);
    cvt(P[bi].w1, P[bi].w1_b, 4 * D_ * D_, stream);
    cvt(P[bi].w2, P[bi].w2_b, 4 * D_ * D_, stream);
  }

  float* OUT1 = (float*)d_out;                           // x_temp1 (N,BT,D)
  float* OUTy = OUT1 + (size_t)N_ * BT_ * D_;            // y_out   (1,BT,D)
  float* OUT2 = OUTy + (size_t)BT_ * D_;                 // x_temp2 (L,BT,D)

  // Three encoder blocks: blk -> X (=x12), ad1 -> OUT2 (x_temp2), ad -> XT (x_temp)
  encoder_block(x, X,    P[0], H, SCR, O, stream);
  encoder_block(X, OUT2, P[2], H, SCR, O, stream);
  encoder_block(X, XT,   P[1], H, SCR, O, stream);

  // Masked pooling -> feat, y_out
  k_pool_mask<<<BT_, N_, 0, stream>>>(mask, pres, rowsum);
  k_counts<<<1, N_, 0, stream>>>(pres, rowsum, counts);
  k_feat<<<dim3(N_, D_ / 256), 256, 0, stream>>>(mask, pres, counts, XT, feat);
  k_yout<<<dim3(BT_ / 8, D_ / 256), 256, 0, stream>>>(pres, feat, OUTy);

  // Cross attention with blk params; q side is broadcast over BT -> compute once
  ln(feat, P[0].ln1g, P[0].ln1b, qln, N_, stream);
  gemm_bf16o(qln, P[0].wi_b, P[0].bi, nullptr, qp, N_, D_, D_, 0, 0, stream);       // Q proj (32 rows)
  ln(x, P[0].ln1g, P[0].ln1b, H, TOK, stream);                                      // kv = LN(x_orig)
  gemm_bf16o(H, P[0].wi_b + (size_t)D_ * D_, P[0].bi + D_, nullptr, SCR,
             TOK, 2 * D_, D_, 0, 0, stream);                                        // K,V
  k_attn_cross<<<dim3(BT_ * NH_, N_ / 16), 32, 0, stream>>>(qp, SCR, mask, O);
  gemm_f32(O, P[0].wo_b, P[0].bo, feat, X2, N_ * BT_, D_, D_, 0, 2, stream);        // + y (row-bcast)
  ln(X2, P[0].ln2g, P[0].ln2b, H, N_ * BT_, stream);
  gemm_bf16o(H, P[0].w1_b, P[0].b1, nullptr, SCR, N_ * BT_, 4 * D_, D_, 1, 0, stream);
  gemm_f32(SCR, P[0].w2_b, P[0].b2, X2, OUT1, N_ * BT_, D_, 4 * D_, 0, 1, stream);  // x_temp1
}
